// depointconv_81501299409047
// MI455X (gfx1250) — compile-verified, matrix operand order checked by reading, not run on hardware
//
#include <hip/hip_runtime.h>
#include <math.h>

// ---------------- constants (from setup_inputs) ----------------
#define BB     4
#define NN     8192
#define DD     64
#define SS     2048
#define KK     32
#define CMID   32
#define COUT   128
#define NGRP   (BB*SS)          // 8192 groups
#define NELEM  (NGRP*KK)        // 262144 grouped elements
#define CNTF   262144.0f
#define EPS    1e-5f

typedef __attribute__((ext_vector_type(16))) __bf16 v16bf;
typedef __attribute__((ext_vector_type(8)))  __bf16 v8bf;
typedef __attribute__((ext_vector_type(8)))  float  v8f;

// ---------------- helpers ----------------
// native RNE f32->bf16 convert (lowers to v_cvt_pk_bf16_f32 on gfx1250)
__device__ __forceinline__ __bf16 tobf(float f) { return (__bf16)f; }

// Per-lane K index for 16-bit A/B WMMA operand layouts (16x32 / 32x16):
// lanes 0-15 hold K in {0..7, 16..23}; lanes 16-31 hold K in {8..15, 24..31};
// vector elements 0..7 map to the first run in order, 8..15 to the second.
__device__ __forceinline__ int wmma_k16(int lane, int t) {
    int base = (lane < 16) ? 0 : 8;
    return base + ((t >= 8) ? 16 : 0) + (t & 7);
}
// BN stats for h-path channel c, derived from coordinate moments in stats[0..8]
__device__ __forceinline__ void hstats(const float* stats, const float* W_h,
                                       const float* b_h, int c,
                                       float& mu, float& inv_sd) {
    float w0 = W_h[c*3+0], w1 = W_h[c*3+1], w2 = W_h[c*3+2], bc = b_h[c];
    const float ic = 1.0f / CNTF;
    float m1 = (w0*stats[0] + w1*stats[1] + w2*stats[2]) * ic;       // E[W.g]
    float e2 = (w0*w0*stats[3] + w1*w1*stats[4] + w2*w2*stats[5]
              + 2.f*(w0*w1*stats[6] + w0*w2*stats[7] + w1*w2*stats[8])) * ic; // E[(W.g)^2]
    mu = m1 + bc;
    float var = e2 + 2.f*bc*m1 + bc*bc - mu*mu;
    inv_sd = rsqrtf(var + EPS);
}

// ---------------- K0: zero stats region ----------------
__global__ void k_zero(float* stats) { stats[threadIdx.x] = 0.f; }

// ---------------- Kprep: W_out f32 -> bf16 ----------------
__global__ void k_prep(const float* __restrict__ W_out, __bf16* __restrict__ woutb) {
    int i = blockIdx.x * 256 + threadIdx.x;
    if (i < COUT * DD * KK) woutb[i] = tobf(W_out[i]);
}

// ---------------- K1: farthest point sampling (block per batch) ----------------
__global__ __launch_bounds__(256) void k_fps(const float* __restrict__ xyz, int* __restrict__ fps) {
    __shared__ float dist[NN];
    __shared__ float rv[256];
    __shared__ int   ri[256];
    __shared__ float cent[3];
    __shared__ int   curf;
    const int b = blockIdx.x, tid = threadIdx.x;
    const float* X = xyz + (size_t)b * 3 * NN;
    for (int n = tid; n < NN; n += 256) dist[n] = 1e10f;
    if (tid == 0) curf = 0;
    __syncthreads();
    for (int it = 0; it < SS; it++) {
        if (tid == 0) {
            int f = curf;
            fps[b*SS + it] = f;
            cent[0] = X[f]; cent[1] = X[NN + f]; cent[2] = X[2*NN + f];
        }
        __syncthreads();
        float cx = cent[0], cy = cent[1], cz = cent[2];
        float bv = -1.f; int bi = 0;
        for (int n = tid; n < NN; n += 256) {
            float dx = X[n] - cx, dy = X[NN + n] - cy, dz = X[2*NN + n] - cz;
            float d2 = dx*dx + dy*dy + dz*dz;
            float dn = fminf(dist[n], d2);
            dist[n] = dn;
            if (dn > bv) { bv = dn; bi = n; }
        }
        rv[tid] = bv; ri[tid] = bi;
        __syncthreads();
        for (int st = 128; st > 0; st >>= 1) {
            if (tid < st) {
                if (rv[tid+st] > rv[tid] || (rv[tid+st] == rv[tid] && ri[tid+st] < ri[tid])) {
                    rv[tid] = rv[tid+st]; ri[tid] = ri[tid+st];
                }
            }
            __syncthreads();
        }
        if (tid == 0) curf = ri[0];
        __syncthreads();
    }
}

// ---------------- K2: ball query + ds + moment reductions (wave per group) ----------------
__global__ __launch_bounds__(256) void k_group(const float* __restrict__ xyz,
                                               const int* __restrict__ fps,
                                               const float* __restrict__ radius_p,
                                               int* __restrict__ idx,
                                               float* __restrict__ ds,
                                               float* __restrict__ stats) {
    __shared__ int sidx[8][KK];
    const int tid = threadIdx.x, wave = tid >> 5, lane = tid & 31;
    const int gw = blockIdx.x * 8 + wave;
    const int b = gw >> 11, s = gw & (SS - 1);
    const float* X = xyz + (size_t)b * 3 * NN;
    const int cpt = fps[b*SS + s];
    const float cx = X[cpt], cy = X[NN + cpt], cz = X[2*NN + cpt];
    const float r = radius_p[0];
    const float r2 = r * r;
    int cnt = 0;
    for (int n0 = 0; n0 < NN; n0 += 32) {
        if (cnt >= KK) break;
        int n = n0 + lane;
        float dx = X[n] - cx, dy = X[NN + n] - cy, dz = X[2*NN + n] - cz;
        float d2 = dx*dx + dy*dy + dz*dz;
        bool pred = d2 <= r2;
        unsigned mask = (unsigned)__ballot(pred);
        int pre = __popc(mask & ((1u << lane) - 1u));
        int slot = cnt + pre;
        if (pred && slot < KK) sidx[wave][slot] = n;
        cnt += __popc(mask);
    }
    if (cnt > KK) cnt = KK;
    __syncthreads();
    int first = (cnt > 0) ? sidx[wave][0] : cpt;
    int nk = (lane < cnt) ? sidx[wave][lane] : first;
    idx[(size_t)gw * KK + lane] = nk;
    // grouped coords with k==0 zeroed (g_mut)
    float gx = X[nk], gy = X[NN + nk], gz = X[2*NN + nk];
    if (lane == 0) { gx = 0.f; gy = 0.f; gz = 0.f; }
    float csum = gx + gy + gz;
    float density = (csum < 1e-10f) ? 1e-10f : csum;
    float inv = 1.0f / density;
    float mx = inv;
    for (int off = 16; off >= 1; off >>= 1) mx = fmaxf(mx, __shfl_xor(mx, off, 32));
    float dsv = inv / mx;
    ds[(size_t)gw * KK + lane] = dsv;
    // moment accumulation for derived BN stats
    float vals[11] = { gx, gy, gz, gx*gx, gy*gy, gz*gz, gx*gy, gx*gz, gy*gz, dsv, dsv*dsv };
    #pragma unroll
    for (int i = 0; i < 11; i++) {
        float v = vals[i];
        for (int off = 16; off >= 1; off >>= 1) v += __shfl_xor(v, off, 32);
        if (lane == 0) atomicAdd(&stats[i], v);
    }
}

// ---------------- K3: n2 pre-activation t2 per element + its stats ----------------
__global__ __launch_bounds__(256) void k_t2(const float* __restrict__ ds,
    const float* W_n1, const float* b_n1, const float* g_n1, const float* be_n1,
    const float* W_n2, const float* b_n2,
    float* __restrict__ t2buf, float* __restrict__ stats) {
    __shared__ float s1[256], s2[256];
    int i = blockIdx.x * 256 + threadIdx.x;
    float a = 0.f, q = 0.f;
    if (i < NELEM) {
        float dsv = ds[i];
        const float ic = 1.0f / CNTF;
        float mds = stats[9] * ic;
        float vds = stats[10] * ic - mds*mds;
        float t2 = b_n2[0];
        #pragma unroll
        for (int c = 0; c < 16; c++) {
            float w  = W_n1[c];
            float m1 = w*mds + b_n1[c];
            float v1 = w*w*vds;
            float h  = g_n1[c] * ((w*dsv + b_n1[c]) - m1) * rsqrtf(v1 + EPS) + be_n1[c];
            t2 += W_n2[c] * fmaxf(h, 0.f);
        }
        t2buf[i] = t2;
        a = t2; q = t2 * t2;
    }
    s1[threadIdx.x] = a; s2[threadIdx.x] = q;
    __syncthreads();
    for (int st = 128; st > 0; st >>= 1) {
        if (threadIdx.x < st) { s1[threadIdx.x] += s1[threadIdx.x + st]; s2[threadIdx.x] += s2[threadIdx.x + st]; }
        __syncthreads();
    }
    if (threadIdx.x == 0) { atomicAdd(&stats[11], s1[0]); atomicAdd(&stats[12], s2[0]); }
}

// ---------------- K3b: t2 -> sigmoid gate (in place) ----------------
__global__ void k_dsc(float* __restrict__ t2buf, const float* __restrict__ stats,
                      const float* g_n2, const float* be_n2) {
    int i = blockIdx.x * 256 + threadIdx.x;
    if (i >= NELEM) return;
    const float ic = 1.0f / CNTF;
    float m2 = stats[11] * ic;
    float v2 = stats[12] * ic - m2*m2;
    float z  = g_n2[0] * (t2buf[i] - m2) * rsqrtf(v2 + EPS) + be_n2[0];
    t2buf[i] = 1.f / (1.f + expf(-z));
}

// ---------------- K4a: per-group matmul mixed = feat . w^T via WMMA bf16 ----------------
__global__ __launch_bounds__(256) void k_mixed(const float* __restrict__ xyz,
                                               const float* __restrict__ points,
                                               const int* __restrict__ idx,
                                               const float* __restrict__ dscbuf,
                                               const float* __restrict__ stats,
                                               const float* W_h, const float* b_h,
                                               const float* g_h, const float* be_h,
                                               __bf16* __restrict__ mixedb) {
    const int tid = blockIdx.x * 256 + threadIdx.x;
    const int gw = tid >> 5, lane = tid & 31;
    if (gw >= NGRP) return;
    const int b = gw >> 11;
    const float* X = xyz + (size_t)b * 3 * NN;
    const float* P = points + (size_t)b * DD * NN;
    const int* I = idx + (size_t)gw * KK;
    const float* DSC = dscbuf + (size_t)gw * KK;

    int nk[16]; float gxv[16], gyv[16], gzv[16], dscv[16];
    #pragma unroll
    for (int t = 0; t < 16; t++) {
        int k = wmma_k16(lane, t);
        int p = I[k];
        nk[t] = p;
        float x = X[p], y = X[NN + p], z = X[2*NN + p];
        if (k == 0) { x = 0.f; y = 0.f; z = 0.f; }
        gxv[t] = x; gyv[t] = y; gzv[t] = z;
        dscv[t] = DSC[k];
    }
    // B operands: w channels (c = m and m+16), BN+relu with derived stats
    const int m = lane & 15;
    v16bf B0{}, B1{};
    {
        int c0 = m, c1 = m + 16;
        float mu0, is0, mu1, is1;
        hstats(stats, W_h, b_h, c0, mu0, is0);
        hstats(stats, W_h, b_h, c1, mu1, is1);
        float w00 = W_h[c0*3], w01 = W_h[c0*3+1], w02 = W_h[c0*3+2], bb0 = b_h[c0];
        float w10 = W_h[c1*3], w11 = W_h[c1*3+1], w12 = W_h[c1*3+2], bb1 = b_h[c1];
        float gg0 = g_h[c0], ee0 = be_h[c0], gg1 = g_h[c1], ee1 = be_h[c1];
        #pragma unroll
        for (int t = 0; t < 16; t++) {
            float t0 = w00*gxv[t] + w01*gyv[t] + w02*gzv[t] + bb0;
            float t1 = w10*gxv[t] + w11*gyv[t] + w12*gzv[t] + bb1;
            float v0 = fmaxf(gg0*(t0 - mu0)*is0 + ee0, 0.f);
            float v1 = fmaxf(gg1*(t1 - mu1)*is1 + ee1, 0.f);
            B0[t] = tobf(v0);
            B1[t] = tobf(v1);
        }
    }
    const int hi = (lane < 16) ? 0 : 8;
    const size_t base = (size_t)gw * (DD * KK);
    #pragma unroll
    for (int dt = 0; dt < 4; dt++) {
        v16bf A{};
        const float* Pr = P + (size_t)(16*dt + m) * NN;
        #pragma unroll
        for (int t = 0; t < 16; t++)
            A[t] = tobf(Pr[nk[t]] * dscv[t]);
        v8f d0{}, d1{};
        d0 = __builtin_amdgcn_wmma_f32_16x16x32_bf16(false, A, false, B0, (short)0, d0, false, false);
        d1 = __builtin_amdgcn_wmma_f32_16x16x32_bf16(false, A, false, B1, (short)0, d1, false, false);
        #pragma unroll
        for (int r = 0; r < 8; r++) {
            int d = 16*dt + r + hi;
            mixedb[base + (size_t)d * KK + m]      = tobf(d0[r]);
            mixedb[base + (size_t)d * KK + 16 + m] = tobf(d1[r]);
        }
    }
}

// ---------------- K4b: Y[128,8192] = W_out[128,2048] x Mixed[2048,8192] via WMMA ----------------
// Operand fetch: per-lane K-subset of the 16-bit layout is two contiguous
// 8-element runs -> two 16B global_load_b128 per operand per K-step.
__global__ __launch_bounds__(256) void k_gemm(const __bf16* __restrict__ woutb,
                                              const __bf16* __restrict__ mixedb,
                                              const float* __restrict__ b_out,
                                              float* __restrict__ ypre,
                                              float* __restrict__ stats) {
    const int wg = blockIdx.x;           // 512 workgroups, 16 columns each
    const int ot = threadIdx.x >> 5;     // o-tile, one per wave
    const int lane = threadIdx.x & 31;
    const int n0 = wg * 16;
    const int m = lane & 15;
    const int hi = (lane < 16) ? 0 : 8;
    const int kb0 = (lane < 16) ? 0 : 8; // K-run base within each 32-wide K block
    const __bf16* Arow = woutb  + (size_t)(16*ot + m) * 2048 + kb0;
    const __bf16* Bcol = mixedb + (size_t)(n0 + m) * 2048 + kb0;
    v8f acc{};
    #pragma unroll 4
    for (int kb = 0; kb < 64; kb++) {
        const int kbase = kb * 32;
        v8bf a_lo = *(const v8bf*)(Arow + kbase);
        v8bf a_hi = *(const v8bf*)(Arow + kbase + 16);
        v8bf b_lo = *(const v8bf*)(Bcol + kbase);
        v8bf b_hi = *(const v8bf*)(Bcol + kbase + 16);
        v16bf A  = __builtin_shufflevector(a_lo, a_hi, 0,1,2,3,4,5,6,7,8,9,10,11,12,13,14,15);
        v16bf Bm = __builtin_shufflevector(b_lo, b_hi, 0,1,2,3,4,5,6,7,8,9,10,11,12,13,14,15);
        acc = __builtin_amdgcn_wmma_f32_16x16x32_bf16(false, A, false, Bm, (short)0, acc, false, false);
    }
    const int b = n0 >> 11;
    const int s = (n0 & (SS - 1)) + m;
    #pragma unroll
    for (int r = 0; r < 8; r++) {
        int o = 16*ot + r + hi;
        float y = acc[r] + b_out[o];
        ypre[((size_t)(b * COUT + o) << 11) + s] = y;
        float a = y, q = y * y;
        for (int off = 8; off >= 1; off >>= 1) {
            a += __shfl_xor(a, off, 16);
            q += __shfl_xor(q, off, 16);
        }
        if (m == 0) { atomicAdd(&stats[16 + 2*o], a); atomicAdd(&stats[17 + 2*o], q); }
    }
}

// ---------------- K5: final BatchNorm ----------------
__global__ void k_bnout(const float* __restrict__ ypre, const float* __restrict__ stats,
                        const float* __restrict__ g_out, const float* __restrict__ be_out,
                        float* __restrict__ out) {
    int i = blockIdx.x * 256 + threadIdx.x;
    if (i >= BB * COUT * SS) return;
    int o = (i >> 11) & (COUT - 1);
    const float ic = 1.0f / (float)(BB * SS);
    float mean = stats[16 + 2*o] * ic;
    float var  = stats[17 + 2*o] * ic - mean*mean;
    out[i] = g_out[o] * (ypre[i] - mean) * rsqrtf(var + EPS) + be_out[o];
}

// ---------------- host launcher ----------------
extern "C" void kernel_launch(void* const* d_in, const int* in_sizes, int n_in,
                              void* d_out, int out_size, void* d_ws, size_t ws_size,
                              hipStream_t stream) {
    (void)in_sizes; (void)n_in; (void)out_size; (void)ws_size;
    const float* xyz   = (const float*)d_in[0];
    const float* points= (const float*)d_in[1];
    const float* W_h   = (const float*)d_in[2];
    const float* b_h   = (const float*)d_in[3];
    const float* g_h   = (const float*)d_in[4];
    const float* be_h  = (const float*)d_in[5];
    const float* W_n1  = (const float*)d_in[6];
    const float* b_n1  = (const float*)d_in[7];
    const float* g_n1  = (const float*)d_in[8];
    const float* be_n1 = (const float*)d_in[9];
    const float* W_n2  = (const float*)d_in[10];
    const float* b_n2  = (const float*)d_in[11];
    const float* g_n2  = (const float*)d_in[12];
    const float* be_n2 = (const float*)d_in[13];
    const float* W_out = (const float*)d_in[14];
    const float* b_out = (const float*)d_in[15];
    const float* g_out = (const float*)d_in[16];
    const float* be_out= (const float*)d_in[17];
    const float* radius= (const float*)d_in[18];
    float* out = (float*)d_out;

    // workspace layout (all regions 16B aligned)
    float* stats = (float*)d_ws;                              // 512 floats (moments + y stats)
    int*   fps   = (int*)(stats + 512);                       // 4*2048
    int*   idx   = fps + BB*SS;                               // 262144
    float* ds    = (float*)(idx + NELEM);                     // 262144
    float* t2buf = ds + NELEM;                                // 262144 (t2 then dsc, in place)
    __bf16* woutb  = (__bf16*)(t2buf + NELEM);                // 262144 bf16
    __bf16* mixedb = woutb + COUT*DD*KK;                      // 8192*2048 bf16
    float* ypre  = (float*)(mixedb + (size_t)NGRP*DD*KK);     // 1048576

    k_zero <<<1, 512, 0, stream>>>(stats);
    k_prep <<<(COUT*DD*KK + 255)/256, 256, 0, stream>>>(W_out, woutb);
    k_fps  <<<BB, 256, 0, stream>>>(xyz, fps);
    k_group<<<NGRP/8, 256, 0, stream>>>(xyz, fps, radius, idx, ds, stats);
    k_t2   <<<NELEM/256, 256, 0, stream>>>(ds, W_n1, b_n1, g_n1, be_n1, W_n2, b_n2, t2buf, stats);
    k_dsc  <<<NELEM/256, 256, 0, stream>>>(t2buf, stats, g_n2, be_n2);
    k_mixed<<<NELEM/256, 256, 0, stream>>>(xyz, points, idx, t2buf, stats,
                                           W_h, b_h, g_h, be_h, mixedb);
    k_gemm <<<NGRP/16, 256, 0, stream>>>(woutb, mixedb, b_out, ypre, stats);
    k_bnout<<<(BB*COUT*SS + 255)/256, 256, 0, stream>>>(ypre, stats, g_out, be_out, out);
}
